// MultiHeadAttention_52329881534544
// MI455X (gfx1250) — compile-verified, matrix operand order checked by reading, not run on hardware
//
#include <hip/hip_runtime.h>

typedef __bf16 bf16;
typedef unsigned int u32;
typedef __attribute__((ext_vector_type(16))) __bf16 v16bf;
typedef __attribute__((ext_vector_type(8)))  float  v8f;
typedef __attribute__((ext_vector_type(4)))  u32    v4u;
typedef __attribute__((ext_vector_type(8)))  int    v8i;
typedef __attribute__((ext_vector_type(4)))  int    v4i;

static constexpr int CB = 2;        // batch
static constexpr int CS = 2048;     // seq
static constexpr int CD = 1024;     // model dim
static constexpr int CH = 16;       // heads
static constexpr int CDK = 64;      // head dim
static constexpr int CM = CB * CS;  // 4096 rows

// ---------------- WMMA helpers (CDNA5 16x16x32 bf16, fp32 accum) ----------------

__device__ __forceinline__ v8f v8f_zero() {
  v8f z;
#pragma unroll
  for (int i = 0; i < 8; ++i) z[i] = 0.0f;
  return z;
}

__device__ __forceinline__ v8f wmma_bf16(v16bf a, v16bf b, v8f c) {
  // (neg_a, A, neg_b, B, c_mod, C, reuse_a, reuse_b)
  return __builtin_amdgcn_wmma_f32_16x16x32_bf16(false, a, false, b, (short)0, c,
                                                 false, false);
}

// 16-bit A-matrix 16x32 fragment (ISA 7.12.2): lanes 0-15 -> M=0..15, K in
// {0..7,16..23}; lanes 16-31 -> same rows, K in {8..15,24..31}. The identical
// pattern loads B (32x16) from a row-major B^T [n,k] buffer (lane -> column n).
__device__ __forceinline__ v16bf frag_ld_bf16(const bf16* __restrict__ base,
                                              int ld, int lane) {
  const int row   = lane & 15;
  const int khalf = (lane & 16) >> 1;  // 0 or 8
  const bf16* r = base + (size_t)row * ld;
  v16bf f;
#pragma unroll
  for (int v = 0; v < 8; ++v) {
    const int k0 = ((v & 4) << 2) + khalf + ((v & 3) << 1);
    f[2 * v]     = r[k0];
    f[2 * v + 1] = r[k0 + 1];
  }
  return f;
}

__device__ __forceinline__ v16bf frag_ld_f32(const float* __restrict__ base,
                                             int ld, int lane) {
  const int row   = lane & 15;
  const int khalf = (lane & 16) >> 1;
  const float* r = base + (size_t)row * ld;
  v16bf f;
#pragma unroll
  for (int v = 0; v < 8; ++v) {
    const int k0 = ((v & 4) << 2) + khalf + ((v & 3) << 1);
    f[2 * v]     = (bf16)r[k0];
    f[2 * v + 1] = (bf16)r[k0 + 1];
  }
  return f;
}

// ---------------- Tensor Data Mover: 2D bf16 tile, 32 elems/row, LDS pad ----------
// D# per CDNA5 ISA. LDS row stride becomes
// 16 DWORDs + 2 DWORDs pad = 72 B = 36 bf16, matching frag_ld_bf16's ld=36.

__device__ __forceinline__ void tdm_load_tile32(u32 lds_byte_addr,
                                                const bf16* gptr,
                                                u32 tile_rows,
                                                u32 tensor_d0, u32 tensor_d1,
                                                u32 stride_elems) {
  const unsigned long long ga = (unsigned long long)gptr;
  v4u g0;
  g0[0] = 1u;                                   // count=1 (valid), user mode
  g0[1] = lds_byte_addr;                        // bits 63:32 lds_addr
  g0[2] = (u32)ga;                              // global_addr[31:0]
  g0[3] = (u32)((ga >> 32) & 0x1FFFFFFu) | (2u << 30);  // addr[56:32], type=2
  v8i g1;
  // data_size=2B(1), pad_enable=1, pad_interval=3 (16 DW), pad_amount=1 (2 DW)
  g1[0] = (int)((1u << 16) | (1u << 20) | (3u << 22) | (1u << 25));
  g1[1] = (int)((tensor_d0 & 0xFFFFu) << 16);   // atomic_barrier_addr=0
  g1[2] = (int)(((tensor_d0 >> 16) & 0xFFFFu) | ((tensor_d1 & 0xFFFFu) << 16));
  g1[3] = (int)(((tensor_d1 >> 16) & 0xFFFFu) | (32u << 16));  // tile_dim0=32
  g1[4] = (int)(tile_rows & 0xFFFFu);           // tile_dim1, tile_dim2=0
  g1[5] = (int)stride_elems;                    // tensor_dim0_stride[31:0]
  g1[6] = 0;                                    // stride hi, dim1_stride lo
  g1[7] = 0;
  const v4i z4 = {0, 0, 0, 0};
  const v8i z8 = {0, 0, 0, 0, 0, 0, 0, 0};
  // 6-arg toolchain form: (g0, g1, g2, g3, extra, cpol)
  __builtin_amdgcn_tensor_load_to_lds(g0, g1, z4, z4, z8, 0);
}

__device__ __forceinline__ u32 lds_addr_of(const void* p) {
  return (u32)(uintptr_t)p;  // low 32 bits of flat LDS address = LDS offset
}

// ---------------- conversion / transpose ----------------

__global__ __launch_bounds__(256) void cvt_bf16_kernel(const float* __restrict__ in,
                                                       bf16* __restrict__ out, int n) {
  const int i = (blockIdx.x * 256 + threadIdx.x) * 4;
  if (i + 3 < n) {
    const float4 v = *(const float4*)(in + i);
    out[i + 0] = (bf16)v.x;
    out[i + 1] = (bf16)v.y;
    out[i + 2] = (bf16)v.z;
    out[i + 3] = (bf16)v.w;
  }
}

// WT[o][i] = (bf16)W[i][o], 1024x1024
__global__ __launch_bounds__(256) void transpose_cvt_kernel(const float* __restrict__ W,
                                                            bf16* __restrict__ WT) {
  __shared__ float tile[32][33];
  const int bx = blockIdx.x * 32, by = blockIdx.y * 32;
  const int tx = threadIdx.x & 31;
  const int ty = (threadIdx.x >> 5) * 4;
#pragma unroll
  for (int r = 0; r < 4; ++r)
    tile[ty + r][tx] = W[(size_t)(by + ty + r) * CD + bx + tx];
  __syncthreads();
#pragma unroll
  for (int r = 0; r < 4; ++r)
    WT[(size_t)(bx + ty + r) * CD + by + tx] = (bf16)tile[tx][ty + r];
}

// ---------------- QKV projection GEMM: C[4096,1024] = A @ W  ----------------
// A bf16 row-major, WT bf16 [out,in]. mode 0 -> dst[B,H,S,DK]; mode 1 -> dst[B,H,DV,S]
// TDM double-buffered tile staging: wave 0 drives the DMA, all waves compute.

__global__ __launch_bounds__(256) void gemm_qkv_kernel(const bf16* __restrict__ A,
                                                       const bf16* __restrict__ WT,
                                                       bf16* __restrict__ dst,
                                                       int mode) {
  __shared__ bf16 As[2][128 * 36];  // TDM pads rows 64B + 8B -> stride 36 bf16
  __shared__ bf16 Bs[2][64 * 36];
  const int t = threadIdx.x;
  const int lane = t & 31, wid = t >> 5;
  const int wm = wid & 3, wn = wid >> 2;  // 4 x 2 waves, 32x32 each
  const int m0 = blockIdx.y * 128;
  const int n0 = blockIdx.x * 64;

  const u32 ldsA[2] = {lds_addr_of(&As[0][0]), lds_addr_of(&As[1][0])};
  const u32 ldsB[2] = {lds_addr_of(&Bs[0][0]), lds_addr_of(&Bs[1][0])};
  const bf16* Arow = A + (size_t)m0 * CD;
  const bf16* Brow = WT + (size_t)n0 * CD;

  v8f acc[2][2];
#pragma unroll
  for (int i = 0; i < 2; ++i)
#pragma unroll
    for (int j = 0; j < 2; ++j) acc[i][j] = v8f_zero();

  if (wid == 0) {
    tdm_load_tile32(ldsA[0], Arow, 128, CD, CM, CD);
    tdm_load_tile32(ldsB[0], Brow, 64, CD, CD, CD);
  }

  int buf = 0;
  for (int kb = 0; kb < CD; kb += 32, buf ^= 1) {
    if (wid == 0) {
      if (kb + 32 < CD) {
        tdm_load_tile32(ldsA[buf ^ 1], Arow + kb + 32, 128, CD, CM, CD);
        tdm_load_tile32(ldsB[buf ^ 1], Brow + kb + 32, 64, CD, CD, CD);
        __builtin_amdgcn_s_wait_tensorcnt(2);  // current pair done; next in flight
      } else {
        __builtin_amdgcn_s_wait_tensorcnt(0);
      }
    }
    __syncthreads();
    const bf16* Ab = As[buf];
    const bf16* Bb = Bs[buf];
    const v16bf a0 = frag_ld_bf16(Ab + (wm * 32)      * 36, 36, lane);
    const v16bf a1 = frag_ld_bf16(Ab + (wm * 32 + 16) * 36, 36, lane);
    const v16bf b0 = frag_ld_bf16(Bb + (wn * 32)      * 36, 36, lane);
    const v16bf b1 = frag_ld_bf16(Bb + (wn * 32 + 16) * 36, 36, lane);
    acc[0][0] = wmma_bf16(a0, b0, acc[0][0]);
    acc[0][1] = wmma_bf16(a0, b1, acc[0][1]);
    acc[1][0] = wmma_bf16(a1, b0, acc[1][0]);
    acc[1][1] = wmma_bf16(a1, b1, acc[1][1]);
    __syncthreads();
  }

  const int col = lane & 15, rb = (lane & 16) >> 1;
#pragma unroll
  for (int i = 0; i < 2; ++i)
#pragma unroll
    for (int j = 0; j < 2; ++j) {
      const int mt = m0 + wm * 32 + i * 16 + rb;
      const int nt = n0 + wn * 32 + j * 16 + col;
      const int h = nt >> 6, dd = nt & 63;
#pragma unroll
      for (int r = 0; r < 8; ++r) {
        const int gm = mt + r;
        const int b = gm >> 11, s = gm & 2047;
        const size_t idx = (mode == 0)
            ? ((size_t)((b * CH + h) * CS + s) * CDK + dd)
            : ((size_t)((b * CH + h) * CDK + dd) * CS + s);
        dst[idx] = (bf16)acc[i][j][r];
      }
    }
}

// ---------------- scores: P[b,h,q,k] = (Q.K^T)/8, masked ----------------

__global__ __launch_bounds__(256) void attn_scores_kernel(const bf16* __restrict__ Q,
                                                          const bf16* __restrict__ K,
                                                          const unsigned char* __restrict__ mask,
                                                          float* __restrict__ P) {
  const int t = threadIdx.x, lane = t & 31, wid = t >> 5;
  const int wm = wid >> 2, wn = wid & 3;  // 2 x 4 waves, 32x32 each
  const int bh = blockIdx.z;
  const int b = bh >> 4;
  const int m0 = blockIdx.y * 64 + wm * 32;
  const int n0 = blockIdx.x * 128 + wn * 32;
  const bf16* Qb = Q + (size_t)bh * CS * CDK;
  const bf16* Kb = K + (size_t)bh * CS * CDK;

  v8f acc[2][2];
#pragma unroll
  for (int i = 0; i < 2; ++i)
#pragma unroll
    for (int j = 0; j < 2; ++j) acc[i][j] = v8f_zero();

#pragma unroll
  for (int kb = 0; kb < CDK; kb += 32) {
    const v16bf a0 = frag_ld_bf16(Qb + (size_t)m0 * CDK + kb, CDK, lane);
    const v16bf a1 = frag_ld_bf16(Qb + (size_t)(m0 + 16) * CDK + kb, CDK, lane);
    const v16bf b0 = frag_ld_bf16(Kb + (size_t)n0 * CDK + kb, CDK, lane);
    const v16bf b1 = frag_ld_bf16(Kb + (size_t)(n0 + 16) * CDK + kb, CDK, lane);
    acc[0][0] = wmma_bf16(a0, b0, acc[0][0]);
    acc[0][1] = wmma_bf16(a0, b1, acc[0][1]);
    acc[1][0] = wmma_bf16(a1, b0, acc[1][0]);
    acc[1][1] = wmma_bf16(a1, b1, acc[1][1]);
  }

  const int col = lane & 15, rb = (lane & 16) >> 1;
  float* Pb = P + (size_t)bh * CS * CS;
  const unsigned char* Mb = mask + (size_t)b * CS * CS;
#pragma unroll
  for (int i = 0; i < 2; ++i)
#pragma unroll
    for (int j = 0; j < 2; ++j)
#pragma unroll
      for (int r = 0; r < 8; ++r) {
        const int gm = m0 + i * 16 + rb + r;
        const int gn = n0 + j * 16 + col;
        const float v = acc[i][j][r] * 0.125f;  // 1/sqrt(64)
        Pb[(size_t)gm * CS + gn] = Mb[(size_t)gm * CS + gn] ? -1e9f : v;
      }
}

// ---------------- row softmax over 2048, in place ----------------

__global__ __launch_bounds__(256) void softmax_kernel(float* __restrict__ P) {
  float* p = P + (size_t)blockIdx.x * CS;
  const int t = threadIdx.x, lane = t & 31, w = t >> 5;
  __shared__ float sm[8];
  float x[8];
  float mx = -3.402823466e38f;
#pragma unroll
  for (int i = 0; i < 8; ++i) { x[i] = p[t + i * 256]; mx = fmaxf(mx, x[i]); }
#pragma unroll
  for (int off = 16; off > 0; off >>= 1) mx = fmaxf(mx, __shfl_xor(mx, off));
  if (lane == 0) sm[w] = mx;
  __syncthreads();
  mx = sm[0];
#pragma unroll
  for (int i = 1; i < 8; ++i) mx = fmaxf(mx, sm[i]);
  float sum = 0.0f;
#pragma unroll
  for (int i = 0; i < 8; ++i) { x[i] = __expf(x[i] - mx); sum += x[i]; }
#pragma unroll
  for (int off = 16; off > 0; off >>= 1) sum += __shfl_xor(sum, off);
  __syncthreads();
  if (lane == 0) sm[w] = sum;
  __syncthreads();
  sum = 0.0f;
#pragma unroll
  for (int i = 0; i < 8; ++i) sum += sm[i];
  const float inv = 1.0f / sum;
#pragma unroll
  for (int i = 0; i < 8; ++i) p[t + i * 256] = x[i] * inv;
}

// ---------------- context: Ctx[b,s,h*64+dv] = P @ V ----------------
// P fp32 [b,h,s,s]; Vt bf16 [b,h,dv,s]; Ctx bf16 [4096,1024]
// Shared V tile staged by TDM (double-buffered); P fragments straight from L2.

__global__ __launch_bounds__(256) void attn_context_kernel(const float* __restrict__ P,
                                                           const bf16* __restrict__ Vt,
                                                           bf16* __restrict__ Ctx) {
  __shared__ bf16 Bs[2][64 * 36];
  const int t = threadIdx.x, lane = t & 31, wid = t >> 5;
  const int bh = blockIdx.y, b = bh >> 4, h = bh & 15;
  const int m0 = blockIdx.x * 128 + wid * 16;  // 8 waves stacked in M
  const float* Pb = P + (size_t)bh * CS * CS;
  const bf16* Vb = Vt + (size_t)bh * CDK * CS;

  const u32 ldsB[2] = {lds_addr_of(&Bs[0][0]), lds_addr_of(&Bs[1][0])};

  v8f acc[4];
#pragma unroll
  for (int j = 0; j < 4; ++j) acc[j] = v8f_zero();

  if (wid == 0) tdm_load_tile32(ldsB[0], Vb, 64, CS, CDK, CS);

  int buf = 0;
  for (int kb = 0; kb < CS; kb += 32, buf ^= 1) {
    if (wid == 0) {
      if (kb + 32 < CS) {
        tdm_load_tile32(ldsB[buf ^ 1], Vb + kb + 32, 64, CS, CDK, CS);
        __builtin_amdgcn_s_wait_tensorcnt(1);
      } else {
        __builtin_amdgcn_s_wait_tensorcnt(0);
      }
    }
    __syncthreads();
    const v16bf a = frag_ld_f32(Pb + (size_t)m0 * CS + kb, CS, lane);
#pragma unroll
    for (int j = 0; j < 4; ++j) {
      const v16bf bf = frag_ld_bf16(Bs[buf] + (j * 16) * 36, 36, lane);
      acc[j] = wmma_bf16(a, bf, acc[j]);
    }
    __syncthreads();
  }

  const int col = lane & 15, rb = (lane & 16) >> 1;
#pragma unroll
  for (int j = 0; j < 4; ++j)
#pragma unroll
    for (int r = 0; r < 8; ++r) {
      const int s = m0 + rb + r;
      const int dv = j * 16 + col;
      Ctx[(size_t)(b * CS + s) * CD + h * CDK + dv] = (bf16)acc[j][r];
    }
}

// ---------------- output projection + residual ----------------

__global__ __launch_bounds__(256) void gemm_out_kernel(const bf16* __restrict__ A,
                                                       const bf16* __restrict__ WT,
                                                       const float* __restrict__ resid,
                                                       float* __restrict__ out) {
  __shared__ bf16 As[2][128 * 36];
  __shared__ bf16 Bs[2][64 * 36];
  const int t = threadIdx.x;
  const int lane = t & 31, wid = t >> 5;
  const int wm = wid & 3, wn = wid >> 2;
  const int m0 = blockIdx.y * 128;
  const int n0 = blockIdx.x * 64;

  const u32 ldsA[2] = {lds_addr_of(&As[0][0]), lds_addr_of(&As[1][0])};
  const u32 ldsB[2] = {lds_addr_of(&Bs[0][0]), lds_addr_of(&Bs[1][0])};
  const bf16* Arow = A + (size_t)m0 * CD;
  const bf16* Brow = WT + (size_t)n0 * CD;

  v8f acc[2][2];
#pragma unroll
  for (int i = 0; i < 2; ++i)
#pragma unroll
    for (int j = 0; j < 2; ++j) acc[i][j] = v8f_zero();

  if (wid == 0) {
    tdm_load_tile32(ldsA[0], Arow, 128, CD, CM, CD);
    tdm_load_tile32(ldsB[0], Brow, 64, CD, CD, CD);
  }

  int buf = 0;
  for (int kb = 0; kb < CD; kb += 32, buf ^= 1) {
    if (wid == 0) {
      if (kb + 32 < CD) {
        tdm_load_tile32(ldsA[buf ^ 1], Arow + kb + 32, 128, CD, CM, CD);
        tdm_load_tile32(ldsB[buf ^ 1], Brow + kb + 32, 64, CD, CD, CD);
        __builtin_amdgcn_s_wait_tensorcnt(2);
      } else {
        __builtin_amdgcn_s_wait_tensorcnt(0);
      }
    }
    __syncthreads();
    const bf16* Ab = As[buf];
    const bf16* Bb = Bs[buf];
    const v16bf a0 = frag_ld_bf16(Ab + (wm * 32)      * 36, 36, lane);
    const v16bf a1 = frag_ld_bf16(Ab + (wm * 32 + 16) * 36, 36, lane);
    const v16bf b0 = frag_ld_bf16(Bb + (wn * 32)      * 36, 36, lane);
    const v16bf b1 = frag_ld_bf16(Bb + (wn * 32 + 16) * 36, 36, lane);
    acc[0][0] = wmma_bf16(a0, b0, acc[0][0]);
    acc[0][1] = wmma_bf16(a0, b1, acc[0][1]);
    acc[1][0] = wmma_bf16(a1, b0, acc[1][0]);
    acc[1][1] = wmma_bf16(a1, b1, acc[1][1]);
    __syncthreads();
  }

  const int col = lane & 15, rb = (lane & 16) >> 1;
#pragma unroll
  for (int i = 0; i < 2; ++i)
#pragma unroll
    for (int j = 0; j < 2; ++j) {
      const int mt = m0 + wm * 32 + i * 16 + rb;
      const int nt = n0 + wn * 32 + j * 16 + col;
#pragma unroll
      for (int r = 0; r < 8; ++r) {
        const size_t idx = (size_t)(mt + r) * CD + nt;
        out[idx] = acc[i][j][r] + resid[idx];
      }
    }
}

// ---------------- in-place LayerNorm over rows of 1024 ----------------

__global__ __launch_bounds__(256) void layernorm_kernel(float* __restrict__ out) {
  float* p = out + (size_t)blockIdx.x * CD;
  const int t = threadIdx.x, lane = t & 31, w = t >> 5;
  __shared__ float smu[8];
  __shared__ float ssq[8];
  float x[4];
  float s = 0.0f, sq = 0.0f;
#pragma unroll
  for (int i = 0; i < 4; ++i) {
    x[i] = p[t + i * 256];
    s += x[i];
    sq += x[i] * x[i];
  }
#pragma unroll
  for (int off = 16; off > 0; off >>= 1) {
    s += __shfl_xor(s, off);
    sq += __shfl_xor(sq, off);
  }
  if (lane == 0) { smu[w] = s; ssq[w] = sq; }
  __syncthreads();
  s = 0.0f; sq = 0.0f;
#pragma unroll
  for (int i = 0; i < 8; ++i) { s += smu[i]; sq += ssq[i]; }
  const float mean = s * (1.0f / CD);
  const float var = sq * (1.0f / CD) - mean * mean;
  const float inv = rsqrtf(var + 1e-5f);
#pragma unroll
  for (int i = 0; i < 4; ++i) p[t + i * 256] = (x[i] - mean) * inv;
}

// ---------------- host orchestration ----------------

extern "C" void kernel_launch(void* const* d_in, const int* in_sizes, int n_in,
                              void* d_out, int out_size, void* d_ws, size_t ws_size,
                              hipStream_t stream) {
  (void)in_sizes; (void)n_in; (void)out_size; (void)ws_size;
  const float* inQ = (const float*)d_in[0];
  const float* inK = (const float*)d_in[1];
  const float* inV = (const float*)d_in[2];
  const unsigned char* mask = (const unsigned char*)d_in[3];
  const float* Wq = (const float*)d_in[4];
  const float* Wk = (const float*)d_in[5];
  const float* Wv = (const float*)d_in[6];
  const float* Wo = (const float*)d_in[7];

  float* out = (float*)d_out;                          // [B,S,D]
  float* attn = out + (size_t)CB * CS * CD;            // [B,H,S,S]

  // workspace layout (bf16 elements), total 64 MB
  bf16* ws  = (bf16*)d_ws;
  bf16* Xq  = ws;                       // 4M
  bf16* Xk  = Xq + (size_t)CM * CD;     // 4M
  bf16* Xv  = Xk + (size_t)CM * CD;     // 4M
  bf16* WqT = Xv + (size_t)CM * CD;     // 1M each
  bf16* WkT = WqT + (size_t)CD * CD;
  bf16* WvT = WkT + (size_t)CD * CD;
  bf16* WoT = WvT + (size_t)CD * CD;
  bf16* Qh  = WoT + (size_t)CD * CD;    // [B,H,S,DK]
  bf16* Kh  = Qh + (size_t)CM * CD;     // [B,H,S,DK]
  bf16* Vt  = Kh + (size_t)CM * CD;     // [B,H,DV,S]
  bf16* Ctx = Vt + (size_t)CM * CD;     // [4096,1024]

  const int nX = CM * CD;  // 4194304

  cvt_bf16_kernel<<<nX / 1024, 256, 0, stream>>>(inQ, Xq, nX);
  cvt_bf16_kernel<<<nX / 1024, 256, 0, stream>>>(inK, Xk, nX);
  cvt_bf16_kernel<<<nX / 1024, 256, 0, stream>>>(inV, Xv, nX);

  transpose_cvt_kernel<<<dim3(32, 32), 256, 0, stream>>>(Wq, WqT);
  transpose_cvt_kernel<<<dim3(32, 32), 256, 0, stream>>>(Wk, WkT);
  transpose_cvt_kernel<<<dim3(32, 32), 256, 0, stream>>>(Wv, WvT);
  transpose_cvt_kernel<<<dim3(32, 32), 256, 0, stream>>>(Wo, WoT);

  // projections: grid = (N/64, M/128)
  gemm_qkv_kernel<<<dim3(16, 32), 256, 0, stream>>>(Xq, WqT, Qh, 0);
  gemm_qkv_kernel<<<dim3(16, 32), 256, 0, stream>>>(Xk, WkT, Kh, 0);
  gemm_qkv_kernel<<<dim3(16, 32), 256, 0, stream>>>(Xv, WvT, Vt, 1);

  // scores: grid = (S/128, S/64, B*H)
  attn_scores_kernel<<<dim3(16, 32, CB * CH), 256, 0, stream>>>(Qh, Kh, mask, attn);

  // softmax: one block per row
  softmax_kernel<<<CB * CH * CS, 256, 0, stream>>>(attn);

  // context: grid = (S/128, B*H)
  attn_context_kernel<<<dim3(16, CB * CH), 256, 0, stream>>>(attn, Vt, Ctx);

  // output projection + residual
  gemm_out_kernel<<<dim3(16, 32), 256, 0, stream>>>(Ctx, WoT, inQ, out);

  // in-place LayerNorm
  layernorm_kernel<<<CM, 256, 0, stream>>>(out);
}
